// Stage2_ObjectReasoner_Deformable_17755394802029
// MI455X (gfx1250) — matrix operand order repaired and checked by reading, not compiled, not run on hardware
//
#include <hip/hip_runtime.h>
#include <hip/hip_bf16.h>
#include <stdint.h>
#include <stddef.h>

// ---------------------------------------------------------------------------
// Types for CDNA5 WMMA (wave32)
// ---------------------------------------------------------------------------
typedef __attribute__((ext_vector_type(16))) __bf16 v16bf;
typedef __attribute__((ext_vector_type(16))) short  v16s;
typedef __attribute__((ext_vector_type(8)))  short  v8s;
typedef __attribute__((ext_vector_type(8)))  float  v8f;

#define DMODEL 256
#define NQUERY 300
#define NHEADS 8
#define HEADD  32
#define NBATCH 16
#define NPAD   320   // attention K padded to a multiple of 32

// fp32 -> bf16 round-to-nearest-even
__device__ __forceinline__ unsigned short f2bf(float f) {
  unsigned int u = __float_as_uint(f);
  unsigned int r = (u + 0x7FFFu + ((u >> 16) & 1u)) >> 16;
  return (unsigned short)r;
}

// ---------------------------------------------------------------------------
// Batched WMMA GEMM:  C[z] = A[z] (M x K, bf16) * Wb[z]^T (N x K, bf16) + bias
// Block = 256 threads (8 waves); tile 128(M) x 64(N); wave = 16 x 64 strip.
// The full 64 x K B panel is staged in LDS ONCE (<=128KB of the 320KB WGP LDS),
// so the K loop has no barriers. M-tail rows are CLAMPED, not masked: an A row
// only feeds its own C row, and OOB rows are never stored, so the inner loop
// stays straight-line (no exec-mask divergence).
// Fragment layouts (A and B share one contiguous k order):
//   A: lane = row (lane&15), (lane>>4)*16 = K base, 16 contiguous bf16
//   B: lane = col (lane&15), (lane>>4)*16 = K base, 16 contiguous bf16
//   C: lane = col, VGPR j = row j (lanes<16) / row 8+j (lanes>=16)
// flags: bit0 = relu, bit1 = store bf16
// ---------------------------------------------------------------------------
__global__ void __launch_bounds__(256)
wmma_gemm(const unsigned short* __restrict__ A, const unsigned short* __restrict__ Wb,
          const float* __restrict__ bias, void* __restrict__ Cv,
          int M, int N, int K, int ldc,
          long long strideA, long long strideW, long long strideC,
          int flags) {
  extern __shared__ __align__(64) unsigned short ldsB[];  // 64 x K bf16

  A  += (long long)blockIdx.z * strideA;
  Wb += (long long)blockIdx.z * strideW;

  const int n0 = blockIdx.x * 64;
  const int m0 = blockIdx.y * 128;

  // ---- cooperative B panel load: Wb[n0+c, 0..K) -> ldsB[c*K ..] (once)
  {
    const int c   = threadIdx.x >> 2;   // 0..63
    const int sub = threadIdx.x & 3;    // 0..3
    const int n   = n0 + c;
    const unsigned short* wr = Wb + (size_t)n * K;
    for (int k = sub * 8; k < K; k += 32) {
      v8s val = {};
      if (n < N) val = *(const v8s*)(wr + k);
      *(v8s*)(&ldsB[(size_t)c * K + k]) = val;
    }
  }
  __syncthreads();

  const int wave  = threadIdx.x >> 5;
  const int lane  = threadIdx.x & 31;
  const int l15   = lane & 15;
  const int khalf = lane >> 4;
  const int arow  = m0 + wave * 16 + l15;
  const int arowc = (arow < M) ? arow : (M - 1);   // clamp: no divergence
  const unsigned short* ap = A + (size_t)arowc * K + khalf * 16;

  v8f acc[4] = {};

  for (int k0 = 0; k0 < K; k0 += 32) {
    // A fragment: two contiguous 16B bf16 loads, no conversion, no branches
    v8s lo = *(const v8s*)(ap + k0);
    v8s hi = *(const v8s*)(ap + k0 + 8);
    __builtin_prefetch(ap + k0 + 32, 0, 3);   // speculative; OOB is dropped
    v16s as;
#pragma unroll
    for (int i = 0; i < 8; ++i) { as[i] = lo[i]; as[8 + i] = hi[i]; }
    const v16bf afrag = __builtin_bit_cast(v16bf, as);

#pragma unroll
    for (int ns = 0; ns < 4; ++ns) {
      v16s bs = *(const v16s*)(&ldsB[(size_t)(ns * 16 + l15) * K + k0 + khalf * 16]);
      acc[ns] = __builtin_amdgcn_wmma_f32_16x16x32_bf16(
          false, afrag, false, __builtin_bit_cast(v16bf, bs),
          (short)0, acc[ns], false, false);
    }
  }

  // ---- store
  float*          Cf = (float*)Cv + (long long)blockIdx.z * strideC;
  unsigned short* Cb = (unsigned short*)Cv + (long long)blockIdx.z * strideC;
#pragma unroll
  for (int ns = 0; ns < 4; ++ns) {
    const int col = n0 + ns * 16 + l15;
    if (col >= N) continue;
    const float bv = bias ? bias[col] : 0.f;
#pragma unroll
    for (int j = 0; j < 8; ++j) {
      const int row = m0 + wave * 16 + (khalf ? 8 + j : j);
      if (row < M) {
        float v = acc[ns][j] + bv;
        if (flags & 1) v = fmaxf(v, 0.f);
        if (flags & 2) Cb[(size_t)row * ldc + col] = f2bf(v);
        else           Cf[(size_t)row * ldc + col] = v;
      }
    }
  }
}

// ---------------------------------------------------------------------------
// Elementwise helpers
// ---------------------------------------------------------------------------
__global__ void cvt_bf16_kernel(const float* __restrict__ src,
                                unsigned short* __restrict__ dst, long long n) {
  long long i = (long long)blockIdx.x * blockDim.x + threadIdx.x;
  if (i < n) dst[i] = f2bf(src[i]);
}

__global__ void bcast_q_kernel(const float* __restrict__ oq, float* __restrict__ q,
                               unsigned short* __restrict__ qbf, int total) {
  int i = blockIdx.x * blockDim.x + threadIdx.x;
  if (i >= total) return;
  int row = i >> 8;
  int d   = i & 255;
  int n   = row % NQUERY;
  float v = oq[n * DMODEL + d];
  q[i]   = v;
  qbf[i] = f2bf(v);
}

// qkv (M,768) fp32 -> qh bf16 (128,NQ,32), kh bf16 (128,NQ,32), vT bf16 (128,32,NPAD; zero-padded)
__global__ void split_heads_kernel(const float* __restrict__ qkv,
                                   unsigned short* __restrict__ qh,
                                   unsigned short* __restrict__ khbf,
                                   unsigned short* __restrict__ vTbf) {
  int i = blockIdx.x * blockDim.x + threadIdx.x;   // over 128*NPAD*32
  int d  = i & 31;
  int n  = (i >> 5) % NPAD;
  int bh = i / (NPAD * 32);
  int b = bh >> 3, h = bh & 7;
  if (n < NQUERY) {
    const float* src = qkv + (size_t)(b * NQUERY + n) * 768;
    qh[((size_t)bh * NQUERY + n) * HEADD + d]   = f2bf(src[h * HEADD + d]);
    khbf[((size_t)bh * NQUERY + n) * HEADD + d] = f2bf(src[DMODEL + h * HEADD + d]);
    vTbf[(size_t)bh * HEADD * NPAD + d * NPAD + n] = f2bf(src[2 * DMODEL + h * HEADD + d]);
  } else {
    vTbf[(size_t)bh * HEADD * NPAD + d * NPAD + n] = 0;
  }
}

// ahead (128,NQ,32) fp32 -> merged bf16 (M,256)
__global__ void merge_heads_kernel(const float* __restrict__ ahead,
                                   unsigned short* __restrict__ merged) {
  int i = blockIdx.x * blockDim.x + threadIdx.x;   // over 128*NQ*32
  int d  = i & 31;
  int n  = (i >> 5) % NQUERY;
  int bh = i / (NQUERY * 32);
  int b = bh >> 3, h = bh & 7;
  merged[(size_t)(b * NQUERY + n) * DMODEL + h * HEADD + d] =
      f2bf(ahead[((size_t)bh * NQUERY + n) * HEADD + d]);
}

// row softmax over 300 valid cols; reads fp32 scores, writes bf16 att (zeros in pad cols)
__global__ void softmax_kernel(const float* __restrict__ s, unsigned short* __restrict__ o,
                               int rows, float scale) {
  int row  = blockIdx.x * 8 + (threadIdx.x >> 5);
  int lane = threadIdx.x & 31;
  if (row >= rows) return;
  const float* p = s + (size_t)row * NPAD;
  unsigned short* q = o + (size_t)row * NPAD;
  float v[10];
  float mx = -3.0e38f;
#pragma unroll
  for (int i = 0; i < 10; ++i) {
    int c = lane + i * 32;
    v[i] = (c < NQUERY) ? p[c] * scale : -3.0e38f;
    mx = fmaxf(mx, v[i]);
  }
#pragma unroll
  for (int off = 16; off >= 1; off >>= 1) mx = fmaxf(mx, __shfl_xor(mx, off, 32));
  float sum = 0.f;
#pragma unroll
  for (int i = 0; i < 10; ++i) {
    int c = lane + i * 32;
    float e = (c < NQUERY) ? __expf(v[i] - mx) : 0.f;
    v[i] = e; sum += e;
  }
#pragma unroll
  for (int off = 16; off >= 1; off >>= 1) sum += __shfl_xor(sum, off, 32);
  float inv = 1.f / sum;
#pragma unroll
  for (int i = 0; i < 10; ++i) {
    int c = lane + i * 32;
    q[c] = (c < NQUERY) ? f2bf(v[i] * inv) : (unsigned short)0;
  }
}

// out = LayerNorm(x + dlt)*g + b ; fp32 out + optional bf16 copy; wave per row
__global__ void ln_kernel(const float* __restrict__ x, const float* __restrict__ dlt,
                          const float* __restrict__ g, const float* __restrict__ b,
                          float* __restrict__ out, unsigned short* __restrict__ outbf,
                          int rows) {
  int row  = blockIdx.x * 8 + (threadIdx.x >> 5);
  int lane = threadIdx.x & 31;
  if (row >= rows) return;
  const float* xr = x + (size_t)row * DMODEL;
  const float* dr = dlt + (size_t)row * DMODEL;
  float v[8];
  float sum = 0.f;
#pragma unroll
  for (int i = 0; i < 8; ++i) {
    int c = lane + i * 32;
    v[i] = xr[c] + dr[c];
    sum += v[i];
  }
#pragma unroll
  for (int off = 16; off >= 1; off >>= 1) sum += __shfl_xor(sum, off, 32);
  float mean = sum * (1.0f / DMODEL);
  float sq = 0.f;
#pragma unroll
  for (int i = 0; i < 8; ++i) { float t = v[i] - mean; sq += t * t; }
#pragma unroll
  for (int off = 16; off >= 1; off >>= 1) sq += __shfl_xor(sq, off, 32);
  float inv = rsqrtf(sq * (1.0f / DMODEL) + 1e-5f);
  float* orow = out + (size_t)row * DMODEL;
#pragma unroll
  for (int i = 0; i < 8; ++i) {
    int c = lane + i * 32;
    float r = (v[i] - mean) * inv * g[c] + b[c];
    orow[c] = r;
    if (outbf) outbf[(size_t)row * DMODEL + c] = f2bf(r);
  }
}

// deformable sampling: wave per (b, nq, h); lane = channel d; bf16 output
__global__ void deform_kernel(const float* __restrict__ off, const float* __restrict__ aw,
                              const float* __restrict__ refp, const int* __restrict__ shapes,
                              const float* __restrict__ vbuf, const unsigned char* __restrict__ mask,
                              unsigned short* __restrict__ ca, int L) {
  int wid  = blockIdx.x * 8 + (threadIdx.x >> 5);
  int lane = threadIdx.x & 31;
  int h    = wid & 7;
  int qrow = wid >> 3;           // b*300+nq
  int b    = qrow / NQUERY;
  int nq   = qrow % NQUERY;

  int H = shapes[0], W = shapes[1];
  float rx = refp[nq * 2 + 0];
  float ry = refp[nq * 2 + 1];

  float a[4];
  float mx = -3.0e38f;
#pragma unroll
  for (int p = 0; p < 4; ++p) { a[p] = aw[(size_t)qrow * 32 + h * 4 + p]; mx = fmaxf(mx, a[p]); }
  float asum = 0.f;
#pragma unroll
  for (int p = 0; p < 4; ++p) { a[p] = __expf(a[p] - mx); asum += a[p]; }
  float ainv = 1.f / asum;

  const float* vb = vbuf + (size_t)b * L * DMODEL + h * HEADD + lane;
  const unsigned char* mb = mask + (size_t)b * L;

  float acc = 0.f;
#pragma unroll
  for (int p = 0; p < 4; ++p) {
    float ox = off[(size_t)qrow * 64 + h * 8 + p * 2 + 0];
    float oy = off[(size_t)qrow * 64 + h * 8 + p * 2 + 1];
    float x = (rx + ox / (float)W) * (float)W - 0.5f;
    float y = (ry + oy / (float)H) * (float)H - 0.5f;
    float x0f = floorf(x), y0f = floorf(y);
    float wx1 = x - x0f, wy1 = y - y0f;
    int x0 = (int)x0f, y0 = (int)y0f;
    float wp = a[p] * ainv;
#pragma unroll
    for (int c = 0; c < 4; ++c) {
      int cx = x0 + (c & 1);
      int cy = y0 + (c >> 1);
      float w = ((c & 1) ? wx1 : 1.f - wx1) * ((c >> 1) ? wy1 : 1.f - wy1);
      if (cx >= 0 && cx < W && cy >= 0 && cy < H) {
        int idx = cy * W + cx;
        if (!mb[idx]) acc += wp * w * vb[(size_t)idx * DMODEL];
      }
    }
  }
  ca[(size_t)qrow * DMODEL + h * HEADD + lane] = f2bf(acc);
}

// ---------------------------------------------------------------------------
// Host orchestration
// ---------------------------------------------------------------------------
static inline void* bump(char*& p, size_t bytes) {
  void* r = (void*)p;
  p += (bytes + 255) & ~(size_t)255;
  return r;
}

extern "C" void kernel_launch(void* const* d_in, const int* in_sizes, int n_in,
                              void* d_out, int out_size, void* d_ws, size_t ws_size,
                              hipStream_t stream) {
  (void)n_in; (void)out_size; (void)ws_size;

  const float*         fused  = (const float*)d_in[0];
  const unsigned char* maskp  = (const unsigned char*)d_in[1];
  const int*           shapes = (const int*)d_in[2];
  const float*         oq     = (const float*)d_in[3];
  const float*         refp   = (const float*)d_in[4];
  const float* in_w  = (const float*)d_in[5];  const float* in_b  = (const float*)d_in[6];
  const float* out_w = (const float*)d_in[7];  const float* out_b = (const float*)d_in[8];
  const float* ln1g  = (const float*)d_in[9];  const float* ln1b  = (const float*)d_in[10];
  const float* ln2g  = (const float*)d_in[11]; const float* ln2b  = (const float*)d_in[12];
  const float* ln3g  = (const float*)d_in[13]; const float* ln3b  = (const float*)d_in[14];
  const float* off_w = (const float*)d_in[15]; const float* off_b = (const float*)d_in[16];
  const float* aw_w  = (const float*)d_in[17]; const float* aw_b  = (const float*)d_in[18];
  const float* v_w   = (const float*)d_in[19]; const float* v_b   = (const float*)d_in[20];
  const float* o_w   = (const float*)d_in[21]; const float* o_b   = (const float*)d_in[22];
  const float* f1_w  = (const float*)d_in[23]; const float* f1_b  = (const float*)d_in[24];
  const float* f2_w  = (const float*)d_in[25]; const float* f2_b  = (const float*)d_in[26];

  const int BL = in_sizes[1];        // B * L
  const int L  = BL / NBATCH;
  const int M  = NBATCH * NQUERY;    // 4800
  const int BH = NBATCH * NHEADS;    // 128

  char* wp = (char*)d_ws;
  unsigned short* w_in  = (unsigned short*)bump(wp, (size_t)768 * 256 * 2);
  unsigned short* w_out = (unsigned short*)bump(wp, (size_t)256 * 256 * 2);
  unsigned short* w_off = (unsigned short*)bump(wp, (size_t)64  * 256 * 2);
  unsigned short* w_aw  = (unsigned short*)bump(wp, (size_t)32  * 256 * 2);
  unsigned short* w_v   = (unsigned short*)bump(wp, (size_t)256 * 256 * 2);
  unsigned short* w_o   = (unsigned short*)bump(wp, (size_t)256 * 256 * 2);
  unsigned short* w_f1  = (unsigned short*)bump(wp, (size_t)1024 * 256 * 2);
  unsigned short* w_f2  = (unsigned short*)bump(wp, (size_t)256 * 1024 * 2);
  unsigned short* fusedbf = (unsigned short*)bump(wp, (size_t)BL * DMODEL * 2);
  float* vbuf   = (float*)bump(wp, (size_t)BL * DMODEL * 4);
  float* qbuf   = (float*)bump(wp, (size_t)M * DMODEL * 4);
  unsigned short* qbf = (unsigned short*)bump(wp, (size_t)M * DMODEL * 2);
  float* tmp    = (float*)bump(wp, (size_t)M * 768 * 4);           // qkv out
  unsigned short* tmpbf = (unsigned short*)bump(wp, (size_t)M * 1024 * 2);  // ffn1 out (bf16)
  unsigned short* qh   = (unsigned short*)bump(wp, (size_t)BH * NQUERY * HEADD * 2);
  unsigned short* khbf = (unsigned short*)bump(wp, (size_t)BH * NQUERY * HEADD * 2);
  unsigned short* vTbf = (unsigned short*)bump(wp, (size_t)BH * HEADD * NPAD * 2);
  float* att    = (float*)bump(wp, (size_t)BH * NQUERY * NPAD * 4);
  unsigned short* attbf = (unsigned short*)bump(wp, (size_t)BH * NQUERY * NPAD * 2);
  float* ahead  = (float*)bump(wp, (size_t)BH * NQUERY * HEADD * 4);
  unsigned short* mergedbf = (unsigned short*)bump(wp, (size_t)M * DMODEL * 2);
  float* delta  = (float*)bump(wp, (size_t)M * DMODEL * 4);
  unsigned short* cabbf = (unsigned short*)bump(wp, (size_t)M * DMODEL * 2);
  float* offb   = (float*)bump(wp, (size_t)M * 64 * 4);
  float* awb    = (float*)bump(wp, (size_t)M * 32 * 4);

  const float attn_scale = 0.17677669529663687f;  // 1/sqrt(32)

#define CVT(src, dst, n) \
  cvt_bf16_kernel<<<(int)(((n) + 255) / 256), 256, 0, stream>>>((src), (dst), (long long)(n))
  CVT(in_w,  w_in,  768 * 256);
  CVT(out_w, w_out, 256 * 256);
  CVT(off_w, w_off, 64 * 256);
  CVT(aw_w,  w_aw,  32 * 256);
  CVT(v_w,   w_v,   256 * 256);
  CVT(o_w,   w_o,   256 * 256);
  CVT(f1_w,  w_f1,  1024 * 256);
  CVT(f2_w,  w_f2,  256 * 1024);
  CVT(fused, fusedbf, (long long)BL * DMODEL);
#undef CVT

  bcast_q_kernel<<<(M * DMODEL) / 256, 256, 0, stream>>>(oq, qbuf, qbf, M * DMODEL);

  // value projection of memory tokens: query-independent, computed ONCE for both layers
  wmma_gemm<<<dim3(4, (BL + 127) / 128, 1), 256, 64 * 256 * 2, stream>>>(
      fusedbf, w_v, v_b, vbuf, BL, 256, 256, 256, 0, 0, 0, 0);

  const int mg = (M + 127) / 128;   // 38 M-tiles

  for (int layer = 0; layer < 2; ++layer) {
    // ---- self attention ----
    wmma_gemm<<<dim3(12, mg, 1), 256, 64 * 256 * 2, stream>>>(
        qbf, w_in, in_b, tmp, M, 768, 256, 768, 0, 0, 0, 0);
    split_heads_kernel<<<(BH * NPAD * HEADD) / 256, 256, 0, stream>>>(tmp, qh, khbf, vTbf);

    wmma_gemm<<<dim3(5, 3, BH), 256, 64 * 32 * 2, stream>>>(
        qh, khbf, nullptr, att, NQUERY, NQUERY, HEADD, NPAD,
        (long long)NQUERY * HEADD, (long long)NQUERY * HEADD, (long long)NQUERY * NPAD, 0);
    softmax_kernel<<<(BH * NQUERY) / 8, 256, 0, stream>>>(att, attbf, BH * NQUERY, attn_scale);

    wmma_gemm<<<dim3(1, 3, BH), 256, 64 * NPAD * 2, stream>>>(
        attbf, vTbf, nullptr, ahead, NQUERY, HEADD, NPAD, HEADD,
        (long long)NQUERY * NPAD, (long long)HEADD * NPAD, (long long)NQUERY * HEADD, 0);
    merge_heads_kernel<<<(BH * NQUERY * HEADD) / 256, 256, 0, stream>>>(ahead, mergedbf);

    wmma_gemm<<<dim3(4, mg, 1), 256, 64 * 256 * 2, stream>>>(
        mergedbf, w_out, out_b, delta, M, 256, 256, 256, 0, 0, 0, 0);
    ln_kernel<<<(M + 7) / 8, 256, 0, stream>>>(qbuf, delta, ln1g, ln1b, qbuf, qbf, M);

    // ---- deformable cross attention ----
    wmma_gemm<<<dim3(1, mg, 1), 256, 64 * 256 * 2, stream>>>(
        qbf, w_off, off_b, offb, M, 64, 256, 64, 0, 0, 0, 0);
    wmma_gemm<<<dim3(1, mg, 1), 256, 64 * 256 * 2, stream>>>(
        qbf, w_aw, aw_b, awb, M, 32, 256, 32, 0, 0, 0, 0);
    deform_kernel<<<M, 256, 0, stream>>>(offb, awb, refp, shapes, vbuf, maskp, cabbf, L);
    wmma_gemm<<<dim3(4, mg, 1), 256, 64 * 256 * 2, stream>>>(
        cabbf, w_o, o_b, delta, M, 256, 256, 256, 0, 0, 0, 0);
    ln_kernel<<<(M + 7) / 8, 256, 0, stream>>>(qbuf, delta, ln2g, ln2b, qbuf, qbf, M);

    // ---- FFN ----  (ffn1 stores bf16 directly so ffn2 reads bf16 A)
    wmma_gemm<<<dim3(16, mg, 1), 256, 64 * 256 * 2, stream>>>(
        qbf, w_f1, f1_b, tmpbf, M, 1024, 256, 1024, 0, 0, 0, 1 | 2);
    wmma_gemm<<<dim3(4, mg, 1), 256, 64 * 1024 * 2, stream>>>(
        tmpbf, w_f2, f2_b, delta, M, 256, 1024, 256, 0, 0, 0, 0);
    float* lnout = (layer == 1) ? (float*)d_out : qbuf;
    unsigned short* lnoutbf = (layer == 1) ? nullptr : qbf;
    ln_kernel<<<(M + 7) / 8, 256, 0, stream>>>(qbuf, delta, ln3g, ln3b, lnout, lnoutbf, M);
  }
}
